// VisionRetentionChunk_29326036697179
// MI455X (gfx1250) — matrix operand order repaired and checked by reading, not compiled, not run on hardware
//
#include <hip/hip_runtime.h>
#include <hip/hip_bf16.h>

typedef __attribute__((ext_vector_type(16))) _Float16 v16h;
typedef __attribute__((ext_vector_type(8)))  float    v8f;

#define NHEADS 8
#define EMBED  256
#define KD     32
#define HH_    56
#define WW_    56
#define BB_    2
#define NN_    (HH_*WW_)      // 3136
#define ROWS_  (BB_*NN_)      // 6272
#define MT_    (ROWS_/16)     // 392 row tiles
#define QT_    (NN_/16)       // 196 query tiles per (b,h)
#define KT2_   (NN_/32)       // 98 key steps of 32

__device__ __forceinline__ v8f wmma_f16(v16h a, v16h b, v8f c) {
  return __builtin_amdgcn_wmma_f32_16x16x32_f16(false, a, false, b, (short)0, c, false, false);
}

// A fragment 16x32 f16, row-major source with row stride ld (elements).
// lane<16: row=lane, halves {K0..7, K16..23}; lane>=16: halves {K8..15, K24..31}.
__device__ __forceinline__ v16h load_A(const _Float16* base, int ld, int lane) {
  int row = lane & 15, lh = lane >> 4;
  const _Float16* p = base + (size_t)row * ld + lh * 8;
  v16h a;
  #pragma unroll
  for (int i = 0; i < 8; ++i) { a[i] = p[i]; a[8 + i] = p[16 + i]; }
  return a;
}

// B fragment 32x16 f16 where B[k][n] = src[n*ld + k].
// lane holds column lane&15; lane<16: K0..7, K16..23; lane>=16: K8..15, K24..31.
__device__ __forceinline__ v16h load_B(const _Float16* base, size_t ld, int lane) {
  int n = lane & 15, kh = lane >> 4;
  const _Float16* p = base + (size_t)n * ld + kh * 8;
  v16h b;
  #pragma unroll
  for (int i = 0; i < 8; ++i) { b[i] = p[i]; b[8 + i] = p[16 + i]; }
  return b;
}

__global__ void cvt_f16_kernel(const float* __restrict__ in, _Float16* __restrict__ out, int n) {
  int i = blockIdx.x * blockDim.x + threadIdx.x;
  if (i < n) out[i] = (_Float16)in[i];
}

// ---------------------------------------------------------------------------
// Fused QKV projection (WMMA) + RoPE (theta_shift) + q scaling.
// grid = (MT_, 16, 3); block = 32 (one wave). blockIdx.z: 0=q 1=k 2=v.
// q,k stored [b,h,N,kd] f16 (A-side / B-side for QK^T); v stored transposed
// [b,h,kd,N] f16 so PV B-fragments are contiguous.
// ---------------------------------------------------------------------------
__global__ __launch_bounds__(32) void qkv_rope_kernel(
    const _Float16* __restrict__ x16,
    const _Float16* __restrict__ wq16, const _Float16* __restrict__ wk16,
    const _Float16* __restrict__ wv16,
    const float* __restrict__ bq, const float* __restrict__ bk, const float* __restrict__ bv,
    const float* __restrict__ sinp, const float* __restrict__ cosp,
    _Float16* __restrict__ q16, _Float16* __restrict__ k16, _Float16* __restrict__ vT16)
{
  int lane = threadIdx.x & 31;
  int mt = blockIdx.x, nt = blockIdx.y, which = blockIdx.z;
  const _Float16* w16  = (which == 0) ? wq16 : (which == 1) ? wk16 : wv16;
  const float*    bias = (which == 0) ? bq   : (which == 1) ? bk   : bv;

  v8f acc = {};
  #pragma unroll
  for (int kt = 0; kt < 8; ++kt) {
    v16h a = load_A(x16 + (size_t)(mt * 16) * EMBED + kt * 32, EMBED, lane);
    v16h b = load_B(w16 + (size_t)(nt * 16) * EMBED + kt * 32, (size_t)EMBED, lane);
    acc = wmma_f16(a, b, acc);
  }

  int lh  = lane >> 4;
  int col = nt * 16 + (lane & 15);
  float bcol = bias[col];
  int head = col >> 5, d = col & 31;

  #pragma unroll
  for (int j = 0; j < 8; ++j) {
    float v = acc[j] + bcol;
    int gm  = mt * 16 + j + 8 * lh;
    int b_  = gm / NN_;
    int pos = gm % NN_;
    if (which < 2) {
      // rotate_every_two: partner column c^1 lives in lane^1 (same row).
      float partner = __shfl_xor(v, 1, 32);
      float rot = (col & 1) ? partner : -partner;
      float sv = sinp[(size_t)pos * KD + d];
      float cv = cosp[(size_t)pos * KD + d];
      v = v * cv + rot * sv;
      if (which == 0) v *= 0.17677669529663687f;  // kd^-0.5 folded into q
      _Float16* dst = (which == 0) ? q16 : k16;
      dst[((size_t)(b_ * NHEADS + head) * NN_ + pos) * KD + d] = (_Float16)v;
    } else {
      vT16[((size_t)(b_ * NHEADS + head) * KD + d) * NN_ + pos] = (_Float16)v;
    }
  }
}

// ---------------------------------------------------------------------------
// Flash attention: one wave owns a 16-row query strip of one (b,h).
// Streams K/V/mask in 32-key steps, online softmax, P transposed via LDS.
// grid = (QT_, 16); block = 32.
// ---------------------------------------------------------------------------
__global__ __launch_bounds__(32) void flash_kernel(
    const _Float16* __restrict__ q16, const _Float16* __restrict__ k16,
    const _Float16* __restrict__ vT16, const float* __restrict__ mask,
    float* __restrict__ ao)
{
  __shared__ _Float16 pbuf[16 * 32];
  int lane = threadIdx.x & 31;
  int qt = blockIdx.x;
  int bh = blockIdx.y;
  int b_ = bh >> 3, h = bh & 7;
  int lh = lane >> 4, lcol = lane & 15;

  v16h qA = load_A(q16 + ((size_t)bh * NN_ + qt * 16) * KD, KD, lane);

  float m[8], l[8];
  v8f O0 = {}, O1 = {};
  #pragma unroll
  for (int j = 0; j < 8; ++j) { m[j] = -1.0e30f; l[j] = 0.0f; }

  const float* maskbase = mask + (size_t)h * NN_ * NN_ + (size_t)(qt * 16) * NN_;

  for (int kt = 0; kt < KT2_; ++kt) {
    int kb = kt * 32;
    v16h kB0 = load_B(k16 + ((size_t)bh * NN_ + kb) * KD,      (size_t)KD, lane);
    v16h kB1 = load_B(k16 + ((size_t)bh * NN_ + kb + 16) * KD, (size_t)KD, lane);
    v8f zero = {};
    v8f S0 = wmma_f16(qA, kB0, zero);
    v8f S1 = wmma_f16(qA, kB1, zero);

    #pragma unroll
    for (int j = 0; j < 8; ++j) {
      const float* mrow = maskbase + (size_t)(j + 8 * lh) * NN_ + kb;
      float s0 = S0[j] + mrow[lcol];
      float s1 = S1[j] + mrow[16 + lcol];
      if (kt + 1 < KT2_) __builtin_prefetch(mrow + 32 + lcol, 0, 3);

      float t = fmaxf(s0, s1);
      #pragma unroll
      for (int mk = 1; mk < 16; mk <<= 1) t = fmaxf(t, __shfl_xor(t, mk, 32));
      float mnew  = fmaxf(m[j], t);
      float alpha = __expf(m[j] - mnew);
      float p0 = __expf(s0 - mnew);
      float p1 = __expf(s1 - mnew);
      float rs = p0 + p1;
      #pragma unroll
      for (int mk = 1; mk < 16; mk <<= 1) rs += __shfl_xor(rs, mk, 32);
      l[j] = l[j] * alpha + rs;
      m[j] = mnew;
      O0[j] *= alpha;
      O1[j] *= alpha;
      // D-layout -> LDS (row-major 16x32) for transpose into A-layout
      pbuf[(j + 8 * lh) * 32 + lcol]      = (_Float16)p0;
      pbuf[(j + 8 * lh) * 32 + 16 + lcol] = (_Float16)p1;
    }
    __syncthreads();
    v16h pA  = load_A(pbuf, 32, lane);
    v16h vB0 = load_B(vT16 + (size_t)(bh * KD) * NN_ + kb,        (size_t)NN_, lane);
    v16h vB1 = load_B(vT16 + (size_t)(bh * KD + 16) * NN_ + kb,   (size_t)NN_, lane);
    O0 = wmma_f16(pA, vB0, O0);
    O1 = wmma_f16(pA, vB1, O1);
    __syncthreads();
  }

  #pragma unroll
  for (int j = 0; j < 8; ++j) {
    float inv = 1.0f / l[j];
    int pos = qt * 16 + j + 8 * lh;
    size_t o = ((size_t)b_ * NN_ + pos) * EMBED + h * KD;
    ao[o + lcol]      = O0[j] * inv;
    ao[o + 16 + lcol] = O1[j] * inv;
  }
}

// Depthwise 5x5 conv (lepe), accumulated into attention output.
__global__ void lepe_kernel(const float* __restrict__ x, const float* __restrict__ w,
                            const float* __restrict__ bias, float* __restrict__ ao)
{
  int idx = blockIdx.x * blockDim.x + threadIdx.x;
  if (idx >= BB_ * HH_ * WW_ * EMBED) return;
  int c = idx & 255;
  int t = idx >> 8;
  int wx = t % WW_; t /= WW_;
  int hy = t % HH_; int b_ = t / HH_;
  float acc = bias[c];
  #pragma unroll
  for (int dy = 0; dy < 5; ++dy) {
    int yy = hy + dy - 2;
    if (yy < 0 || yy >= HH_) continue;
    #pragma unroll
    for (int dx = 0; dx < 5; ++dx) {
      int xx = wx + dx - 2;
      if (xx < 0 || xx >= WW_) continue;
      acc += x[(((size_t)b_ * HH_ + yy) * WW_ + xx) * EMBED + c] * w[(dy * 5 + dx) * EMBED + c];
    }
  }
  ao[idx] += acc;
}

// Output projection: out = (ao) @ wo^T + bo, WMMA with on-the-fly f32->f16 A.
__global__ __launch_bounds__(32) void proj_kernel(
    const float* __restrict__ ao, const _Float16* __restrict__ wo16,
    const float* __restrict__ bo, float* __restrict__ out)
{
  int lane = threadIdx.x & 31;
  int mt = blockIdx.x, nt = blockIdx.y;
  int row = lane & 15, lh = lane >> 4;
  v8f acc = {};
  #pragma unroll
  for (int kt = 0; kt < 8; ++kt) {
    const float* p = ao + (size_t)(mt * 16 + row) * EMBED + kt * 32 + lh * 8;
    v16h a;
    #pragma unroll
    for (int i = 0; i < 8; ++i) { a[i] = (_Float16)p[i]; a[8 + i] = (_Float16)p[16 + i]; }
    v16h b = load_B(wo16 + (size_t)(nt * 16) * EMBED + kt * 32, (size_t)EMBED, lane);
    acc = wmma_f16(a, b, acc);
  }
  int col = nt * 16 + (lane & 15);
  float bcol = bo[col];
  #pragma unroll
  for (int j = 0; j < 8; ++j) {
    int gm = mt * 16 + j + 8 * lh;
    out[(size_t)gm * EMBED + col] = acc[j] + bcol;
  }
}

extern "C" void kernel_launch(void* const* d_in, const int* in_sizes, int n_in,
                              void* d_out, int out_size, void* d_ws, size_t ws_size,
                              hipStream_t stream) {
  const float* x      = (const float*)d_in[0];
  const float* sinp   = (const float*)d_in[1];
  const float* cosp   = (const float*)d_in[2];
  const float* mask   = (const float*)d_in[3];
  const float* wq     = (const float*)d_in[4];
  const float* bq     = (const float*)d_in[5];
  const float* wk     = (const float*)d_in[6];
  const float* bk     = (const float*)d_in[7];
  const float* wv     = (const float*)d_in[8];
  const float* bv     = (const float*)d_in[9];
  const float* lepe_w = (const float*)d_in[10];
  const float* lepe_b = (const float*)d_in[11];
  const float* wo     = (const float*)d_in[12];
  const float* bo     = (const float*)d_in[13];
  float* out = (float*)d_out;

  const int NX = BB_ * HH_ * WW_ * EMBED;   // 1,605,632
  const int NW = EMBED * EMBED;             // 65,536

  char* ws = (char*)d_ws;
  size_t off = 0;
  auto take = [&](size_t bytes) { char* p = ws + off; off += (bytes + 255) & ~(size_t)255; return p; };
  _Float16* x16  = (_Float16*)take((size_t)NX * 2);
  _Float16* wq16 = (_Float16*)take((size_t)NW * 2);
  _Float16* wk16 = (_Float16*)take((size_t)NW * 2);
  _Float16* wv16 = (_Float16*)take((size_t)NW * 2);
  _Float16* wo16 = (_Float16*)take((size_t)NW * 2);
  _Float16* q16  = (_Float16*)take((size_t)NX * 2);
  _Float16* k16  = (_Float16*)take((size_t)NX * 2);
  _Float16* vT16 = (_Float16*)take((size_t)NX * 2);
  float*    ao   = (float*)take((size_t)NX * 4);
  (void)ws_size; (void)in_sizes; (void)n_in; (void)out_size;

  cvt_f16_kernel<<<(NX + 255) / 256, 256, 0, stream>>>(x,  x16,  NX);
  cvt_f16_kernel<<<(NW + 255) / 256, 256, 0, stream>>>(wq, wq16, NW);
  cvt_f16_kernel<<<(NW + 255) / 256, 256, 0, stream>>>(wk, wk16, NW);
  cvt_f16_kernel<<<(NW + 255) / 256, 256, 0, stream>>>(wv, wv16, NW);
  cvt_f16_kernel<<<(NW + 255) / 256, 256, 0, stream>>>(wo, wo16, NW);

  dim3 gQKV(MT_, 16, 3);
  qkv_rope_kernel<<<gQKV, 32, 0, stream>>>(x16, wq16, wk16, wv16, bq, bk, bv,
                                           sinp, cosp, q16, k16, vT16);

  dim3 gFA(QT_, BB_ * NHEADS);
  flash_kernel<<<gFA, 32, 0, stream>>>(q16, k16, vT16, mask, ao);

  lepe_kernel<<<(NX + 255) / 256, 256, 0, stream>>>(x, lepe_w, lepe_b, ao);

  dim3 gP(MT_, 16);
  proj_kernel<<<gP, 32, 0, stream>>>(ao, wo16, bo, out);
}
